// MambaBlock_90383291777598
// MI455X (gfx1250) — matrix operand order repaired and checked by reading, not compiled, not run on hardware
//
#include <hip/hip_runtime.h>
#include <hip/hip_bf16.h>

// ---------------------------------------------------------------------------
// Mamba block for MI455X (gfx1250): bf16 WMMA GEMMs with double-buffered
// async global->LDS staging (GLOBAL_LOAD_ASYNC_TO_LDS_B128 + s_wait_asynccnt),
// register-resident selective scan.
// B=4, L=1024, DM=768, DI=1536, DS=16, DTR=48, K=4
// ---------------------------------------------------------------------------

#define B_C   4
#define L_C   1024
#define DM_C  768
#define DI_C  1536
#define DS_C  16
#define DTR_C 48
#define DTRP_C 64                  // dt_r K padded to 64 (zero-filled)
#define KC_C  4
#define M_C   (B_C * L_C)          // 4096 rows

typedef __attribute__((ext_vector_type(16))) __bf16 v16bf;
typedef __attribute__((ext_vector_type(8)))  float  v8f;

// issue one 16-byte async global->LDS copy (per active lane)
__device__ __forceinline__ void async_copy16(unsigned lds_addr, const void* gptr) {
    asm volatile("global_load_async_to_lds_b128 %0, %1, off"
                 :: "v"(lds_addr), "v"((unsigned long long)gptr)
                 : "memory");
}

// ---------------------------------------------------------------------------
// generic f32 -> bf16 convert
// ---------------------------------------------------------------------------
__global__ __launch_bounds__(256)
void f32_to_bf16_kernel(const float* __restrict__ in, __bf16* __restrict__ out, int n) {
    int i = blockIdx.x * 256 + threadIdx.x;
    if (i < n) out[i] = (__bf16)in[i];
}

// ---------------------------------------------------------------------------
// fused double RMSNorm: x_copy = rms(h)*w_outer ; x_in = bf16(rms(x_copy)*w_inner)
// one block (256 threads) per row of DM=768 (3 elements per thread)
// ---------------------------------------------------------------------------
__global__ __launch_bounds__(256)
void rmsnorm2_kernel(const float* __restrict__ h,
                     const float* __restrict__ w_outer,
                     const float* __restrict__ w_inner,
                     float* __restrict__ x_copy,
                     __bf16* __restrict__ x_in) {
    const int row = blockIdx.x;
    const float* hr = h + (size_t)row * DM_C;
    __shared__ float red[8];
    __shared__ float total;
    const int tid = threadIdx.x;
    const int lane = tid & 31, wv = tid >> 5;

    float v[3], xc[3];
    float ss = 0.f;
#pragma unroll
    for (int j = 0; j < 3; ++j) { v[j] = hr[tid + j * 256]; ss += v[j] * v[j]; }
#pragma unroll
    for (int off = 16; off; off >>= 1) ss += __shfl_down(ss, off, 32);
    if (lane == 0) red[wv] = ss;
    __syncthreads();
    if (tid == 0) { float t = 0.f; for (int i = 0; i < 8; ++i) t += red[i]; total = t; }
    __syncthreads();
    float rs = rsqrtf(total / (float)DM_C + 1e-5f);

    ss = 0.f;
#pragma unroll
    for (int j = 0; j < 3; ++j) {
        xc[j] = v[j] * rs * w_outer[tid + j * 256];
        x_copy[(size_t)row * DM_C + tid + j * 256] = xc[j];
        ss += xc[j] * xc[j];
    }
#pragma unroll
    for (int off = 16; off; off >>= 1) ss += __shfl_down(ss, off, 32);
    if (lane == 0) red[wv] = ss;
    __syncthreads();
    if (tid == 0) { float t = 0.f; for (int i = 0; i < 8; ++i) t += red[i]; total = t; }
    __syncthreads();
    float rs2 = rsqrtf(total / (float)DM_C + 1e-5f);
#pragma unroll
    for (int j = 0; j < 3; ++j)
        x_in[(size_t)row * DM_C + tid + j * 256] = (__bf16)(xc[j] * rs2 * w_inner[tid + j * 256]);
}

// ---------------------------------------------------------------------------
// bf16 WMMA GEMM: C[M,N] = A[M,K] @ Bw[N,K]^T  (Bw row-major N x K, as stored)
// block = 256 threads = 8 waves; block tile = 128 x (NSUB*16); K tile = 32.
// Double-buffered async global->LDS staging; Kd must be a multiple of 32.
// MODE 0: plain store. MODE 1: C = acc + add0 + add1.
// ---------------------------------------------------------------------------
template <int NSUB, int MODE>
__global__ __launch_bounds__(256)
void gemm_bf16_wmma(const __bf16* __restrict__ A,
                    const __bf16* __restrict__ Bw,
                    float* __restrict__ C,
                    const float* __restrict__ add0,
                    const float* __restrict__ add1,
                    int M, int N, int Kd) {
    constexpr int BM = 128;
    constexpr int BN = NSUB * 16;
    constexpr int A_CHUNKS = (BM * 32) / 8;          // 512 x 16B
    constexpr int B_CHUNKS = (BN * 32) / 8;          // 256 or 320 x 16B
    constexpr int A_CH = A_CHUNKS / 256;             // 2 per thread
    constexpr int B_CH = (B_CHUNKS + 255) / 256;     // 1 or 2 per thread
    constexpr int OPS  = A_CH + B_CH;                // async ops per wave per tile

    __shared__ __align__(16) __bf16 sA[2][BM * 32];
    __shared__ __align__(16) __bf16 sB[2][BN * 32];

    const int tid  = threadIdx.x;
    const int wave = tid >> 5;
    const int lane = tid & 31;
    const int l16  = lane & 15;
    const int hsel = lane >> 4;          // 0: lanes 0-15, 1: lanes 16-31
    const int m0 = blockIdx.x * BM;
    const int n0 = blockIdx.y * BN;

    auto stage = [&](int buf, int k0) {
        const unsigned ldsA0 = (unsigned)(unsigned long long)(void*)&sA[buf][0];
        const unsigned ldsB0 = (unsigned)(unsigned long long)(void*)&sB[buf][0];
#pragma unroll
        for (int c = 0; c < A_CH; ++c) {
            int idx = tid + c * 256;                 // 16B chunk id
            int r = idx >> 2, c8 = (idx & 3) * 8;
            async_copy16(ldsA0 + (unsigned)(r * 32 + c8) * 2,
                         &A[(size_t)(m0 + r) * Kd + k0 + c8]);
        }
#pragma unroll
        for (int c = 0; c < B_CH; ++c) {
            int idx = tid + c * 256;
            if (idx >= B_CHUNKS) idx -= B_CHUNKS;    // wrap: uniform issue count
            int r = idx >> 2, c8 = (idx & 3) * 8;
            async_copy16(ldsB0 + (unsigned)(r * 32 + c8) * 2,
                         &Bw[(size_t)(n0 + r) * Kd + k0 + c8]);
        }
    };

    v8f acc[NSUB];
#pragma unroll
    for (int j = 0; j < NSUB; ++j) acc[j] = v8f{0.f,0.f,0.f,0.f,0.f,0.f,0.f,0.f};

    const int T = Kd >> 5;                           // #K tiles (Kd % 32 == 0)
    stage(0, 0);
    for (int i = 0; i < T; ++i) {
        const int cur = i & 1;
        if (i + 1 < T) {
            stage(cur ^ 1, (i + 1) * 32);
            // next tile's OPS ops are in flight; in-order completion =>
            // asynccnt <= OPS proves tile i's copies are done.
            asm volatile("s_wait_asynccnt %0" :: "n"(OPS) : "memory");
        } else {
            asm volatile("s_wait_asynccnt 0" ::: "memory");
        }
        __syncthreads();

        // A fragment: lane<16 -> row=l16, K {0..15}; lane>=16 -> K {16..31}
        v16bf afrag;
        {
            const __bf16* pr = &sA[cur][(wave * 16 + l16) * 32];
            const int klo = hsel ? 16 : 0;
#pragma unroll
            for (int q = 0; q < 8; ++q) {
                afrag[q]     = pr[klo + q];
                afrag[8 + q] = pr[klo + 8 + q];
            }
        }
#pragma unroll
        for (int j = 0; j < NSUB; ++j) {
            v16bf bfrag;
            const __bf16* pc = &sB[cur][(j * 16 + l16) * 32 + (hsel ? 16 : 0)];
#pragma unroll
            for (int q = 0; q < 16; ++q) bfrag[q] = pc[q];
            acc[j] = __builtin_amdgcn_wmma_f32_16x16x32_bf16(
                false, afrag, false, bfrag, (short)0, acc[j], false, false);
        }
        __syncthreads();                             // tile consumed; safe to overwrite
    }

    // C/D layout: VGPR v -> row v + 8*hsel, col = l16
    const int rbase = m0 + wave * 16 + hsel * 8;
#pragma unroll
    for (int j = 0; j < NSUB; ++j) {
        const int col = n0 + j * 16 + l16;
#pragma unroll
        for (int v = 0; v < 8; ++v) {
            size_t o = (size_t)(rbase + v) * N + col;
            float val = acc[j][v];
            if (MODE == 1) val += add0[o] + add1[o];
            C[o] = val;
        }
    }
}

// ---------------------------------------------------------------------------
// causal depthwise conv (K=4) + bias + SiLU -> u (f32 + bf16)
// ---------------------------------------------------------------------------
__global__ __launch_bounds__(256)
void conv_silu_kernel(const float* __restrict__ xz,
                      const float* __restrict__ conv_w,
                      const float* __restrict__ conv_b,
                      float* __restrict__ u, __bf16* __restrict__ u_bf) {
    const size_t idx = (size_t)blockIdx.x * 256 + threadIdx.x;
    const size_t total = (size_t)B_C * L_C * DI_C;
    if (idx >= total) return;
    const int d  = (int)(idx % DI_C);
    const size_t bl = idx / DI_C;
    const int l  = (int)(bl % L_C);
    const int b  = (int)(bl / L_C);
    float acc = conv_b[d];
#pragma unroll
    for (int k = 0; k < KC_C; ++k) {
        int ls = l + k - (KC_C - 1);
        if (ls >= 0)
            acc += xz[((size_t)(b * L_C + ls)) * (2 * DI_C) + d] * conv_w[d * KC_C + k];
    }
    float s = acc / (1.f + __expf(-acc));   // silu
    u[idx] = s;
    u_bf[idx] = (__bf16)s;
}

// ---------------------------------------------------------------------------
// extract dt_r (first 48 cols of x_dbl) as bf16 [M,64], zero-padded K tail
// ---------------------------------------------------------------------------
__global__ __launch_bounds__(256)
void dtr_extract_pad_kernel(const float* __restrict__ x_dbl, __bf16* __restrict__ dtr) {
    int idx = blockIdx.x * 256 + threadIdx.x;
    const int total = M_C * DTRP_C;
    if (idx >= total) return;
    int m = idx >> 6, j = idx & 63;
    dtr[idx] = (j < DTR_C) ? (__bf16)x_dbl[(size_t)m * (DTR_C + 2 * DS_C) + j]
                           : (__bf16)0.0f;
}

// dt_proj_w (DI x 48) -> bf16 (DI x 64), zero-padded K tail
__global__ __launch_bounds__(256)
void dtproj_pad_kernel(const float* __restrict__ w, __bf16* __restrict__ out) {
    int idx = blockIdx.x * 256 + threadIdx.x;
    const int total = DI_C * DTRP_C;
    if (idx >= total) return;
    int r = idx >> 6, j = idx & 63;
    out[idx] = (j < DTR_C) ? (__bf16)w[(size_t)r * DTR_C + j] : (__bf16)0.0f;
}

// ---------------------------------------------------------------------------
// selective scan: one thread per (b,d) channel, DS=16 state in registers.
// B/C chunk (shared per (b,l) across block) staged via async global->LDS.
// Fuses: softplus, exp recurrence, y = state.C, +u*D, *silu(res), bf16 out.
// ---------------------------------------------------------------------------
#define SCAN_CHUNK 64
__global__ __launch_bounds__(128)
void scan_kernel(const float* __restrict__ delta,   // (B,L,DI) raw
                 const float* __restrict__ x_dbl,   // (B,L,80)
                 const float* __restrict__ u,       // (B,L,DI)
                 const float* __restrict__ xz,      // res at [..., DI:2DI]
                 const float* __restrict__ A_log,   // (DI,DS)
                 const float* __restrict__ Dvec,    // (DI)
                 const float* __restrict__ dt_bias, // (DI)
                 __bf16* __restrict__ ygate) {
    const int tid = threadIdx.x;
    const int c = blockIdx.x * 128 + tid;   // DI%128==0 -> b uniform per block
    const int b = c / DI_C;
    const int d = c % DI_C;

    __shared__ __align__(16) float sBC[SCAN_CHUNK * 32];
    const unsigned ldsBC = (unsigned)(unsigned long long)(void*)&sBC[0];

    float Arow[DS_C];
#pragma unroll
    for (int n = 0; n < DS_C; ++n) Arow[n] = -__expf(A_log[d * DS_C + n]);
    const float Dd = Dvec[d];
    const float bias = dt_bias[d];

    float st[DS_C];
#pragma unroll
    for (int n = 0; n < DS_C; ++n) st[n] = 0.f;

    for (int l0 = 0; l0 < L_C; l0 += SCAN_CHUNK) {
        __syncthreads();   // previous chunk fully consumed before overwrite
        // 64 rows x 32 floats = 512 x 16B chunks; 4 async copies per thread
#pragma unroll
        for (int cc = 0; cc < 4; ++cc) {
            int idx = tid + cc * 128;        // 16B chunk id (8 chunks per row)
            int lo = idx >> 3;
            int j4 = (idx & 7) * 4;
            async_copy16(ldsBC + (unsigned)idx * 16,
                         &x_dbl[((size_t)(b * L_C + l0 + lo)) * (DTR_C + 2 * DS_C)
                                + DTR_C + j4]);
        }
        asm volatile("s_wait_asynccnt 0" ::: "memory");
        __syncthreads();

        for (int lo = 0; lo < SCAN_CHUNK; ++lo) {
            const int l = l0 + lo;
            const size_t base = ((size_t)(b * L_C + l)) * DI_C + d;
            const float raw = delta[base];
            const float uu  = u[base];
            const float dtb = raw + bias;
            const float dts = (dtb > 20.f) ? dtb : log1pf(__expf(dtb)); // softplus
            const float rbu = raw * uu;   // reference uses RAW delta for deltaB_u
            const float* bc = &sBC[lo * 32];
            float y = 0.f;
#pragma unroll
            for (int n = 0; n < DS_C; ++n) {
                float dA = __expf(dts * Arow[n]);
                st[n] = dA * st[n] + rbu * bc[n];
                y += st[n] * bc[DS_C + n];
            }
            const float rr = xz[((size_t)(b * L_C + l)) * (2 * DI_C) + DI_C + d];
            const float g = rr / (1.f + __expf(-rr));
            ygate[base] = (__bf16)((y + uu * Dd) * g);
        }
    }
}

// ---------------------------------------------------------------------------
// host launcher
// ---------------------------------------------------------------------------
extern "C" void kernel_launch(void* const* d_in, const int* in_sizes, int n_in,
                              void* d_out, int out_size, void* d_ws, size_t ws_size,
                              hipStream_t stream) {
    const float* hidden      = (const float*)d_in[0];
    const float* norm_outer  = (const float*)d_in[1];
    const float* norm_inner  = (const float*)d_in[2];
    const float* in_proj_w   = (const float*)d_in[3];
    const float* conv_w      = (const float*)d_in[4];
    const float* conv_b      = (const float*)d_in[5];
    const float* x_proj_w    = (const float*)d_in[6];
    const float* dt_proj_w   = (const float*)d_in[7];
    const float* dt_proj_b   = (const float*)d_in[8];
    const float* A_log       = (const float*)d_in[9];
    const float* Dvec        = (const float*)d_in[10];
    const float* out_proj_w  = (const float*)d_in[11];
    float* out = (float*)d_out;

    // workspace layout
    char* w = (char*)d_ws;
    auto alloc = [&](size_t bytes) -> char* {
        char* p = w;
        w += (bytes + 255) & ~(size_t)255;
        return p;
    };
    float*  x_copy   = (float*) alloc((size_t)M_C * DM_C * 4);
    __bf16* x_in     = (__bf16*)alloc((size_t)M_C * DM_C * 2);
    __bf16* inproj_b = (__bf16*)alloc((size_t)2 * DI_C * DM_C * 2);
    __bf16* xproj_b  = (__bf16*)alloc((size_t)(DTR_C + 2 * DS_C) * DI_C * 2);
    __bf16* dtproj_b = (__bf16*)alloc((size_t)DI_C * DTRP_C * 2);
    __bf16* outproj_b= (__bf16*)alloc((size_t)DM_C * DI_C * 2);
    float*  xz       = (float*) alloc((size_t)M_C * 2 * DI_C * 4);
    float*  u_f      = (float*) alloc((size_t)M_C * DI_C * 4);
    __bf16* u_bf     = (__bf16*)alloc((size_t)M_C * DI_C * 2);
    float*  x_dbl    = (float*) alloc((size_t)M_C * (DTR_C + 2 * DS_C) * 4);
    __bf16* dtr_bf   = (__bf16*)alloc((size_t)M_C * DTRP_C * 2);
    float*  delta    = (float*) alloc((size_t)M_C * DI_C * 4);
    __bf16* ygate    = (__bf16*)alloc((size_t)M_C * DI_C * 2);
    (void)ws_size; (void)n_in; (void)in_sizes; (void)out_size;

    // 1) weight converts to bf16 (dt_proj padded to K=64)
    {
        int n;
        n = 2 * DI_C * DM_C;
        f32_to_bf16_kernel<<<(n + 255) / 256, 256, 0, stream>>>(in_proj_w, inproj_b, n);
        n = (DTR_C + 2 * DS_C) * DI_C;
        f32_to_bf16_kernel<<<(n + 255) / 256, 256, 0, stream>>>(x_proj_w, xproj_b, n);
        n = DI_C * DTRP_C;
        dtproj_pad_kernel<<<(n + 255) / 256, 256, 0, stream>>>(dt_proj_w, dtproj_b);
        n = DM_C * DI_C;
        f32_to_bf16_kernel<<<(n + 255) / 256, 256, 0, stream>>>(out_proj_w, outproj_b, n);
    }

    // 2) fused double RMSNorm
    rmsnorm2_kernel<<<M_C, 256, 0, stream>>>(hidden, norm_outer, norm_inner, x_copy, x_in);

    // 3) GEMM1: x_in (4096x768) x in_proj (3072x768)^T -> xz (4096x3072)
    gemm_bf16_wmma<4, 0><<<dim3(M_C / 128, (2 * DI_C) / 64), 256, 0, stream>>>(
        x_in, inproj_b, xz, nullptr, nullptr, M_C, 2 * DI_C, DM_C);

    // 4) conv + silu -> u
    {
        size_t total = (size_t)M_C * DI_C;
        conv_silu_kernel<<<(unsigned)((total + 255) / 256), 256, 0, stream>>>(
            xz, conv_w, conv_b, u_f, u_bf);
    }

    // 5) GEMM2: u (4096x1536) x x_proj (80x1536)^T -> x_dbl (4096x80)
    gemm_bf16_wmma<5, 0><<<dim3(M_C / 128, 1), 256, 0, stream>>>(
        u_bf, xproj_b, x_dbl, nullptr, nullptr, M_C, DTR_C + 2 * DS_C, DI_C);

    // 6) extract dt_r bf16 (K padded to 64)
    {
        int n = M_C * DTRP_C;
        dtr_extract_pad_kernel<<<(n + 255) / 256, 256, 0, stream>>>(x_dbl, dtr_bf);
    }

    // 7) GEMM3: dt_r (4096x64) x dt_proj (1536x64)^T -> delta (4096x1536)
    gemm_bf16_wmma<4, 0><<<dim3(M_C / 128, DI_C / 64), 256, 0, stream>>>(
        dtr_bf, dtproj_b, delta, nullptr, nullptr, M_C, DI_C, DTRP_C);

    // 8) selective scan (+ u*D, * silu(res)) -> ygate bf16
    scan_kernel<<<(B_C * DI_C) / 128, 128, 0, stream>>>(
        delta, x_dbl, u_f, xz, A_log, Dvec, dt_proj_b, ygate);

    // 9) GEMM4: ygate (4096x1536) x out_proj (768x1536)^T + x_copy + hidden -> out
    gemm_bf16_wmma<4, 1><<<dim3(M_C / 128, DM_C / 64), 256, 0, stream>>>(
        ygate, outproj_b, out, x_copy, hidden, M_C, DM_C, DI_C);
}
